// MoEMLP_17961553232608
// MI455X (gfx1250) — matrix operand order repaired and checked by reading, not compiled, not run on hardware
//
#include <hip/hip_runtime.h>
#include <hip/hip_bf16.h>
#include <math.h>

// ---------------------------------------------------------------------------
// MoE GeGLU MLP + top-1 LoRA experts, fused for MI455X (gfx1250, wave32).
// Memory-bound: stream 384MB of fp32 weights once, convert to bf16 in-flight,
// accumulate with v_wmma_f32_16x16x32_bf16 into persistent VGPR accumulators.
// ---------------------------------------------------------------------------

typedef __attribute__((ext_vector_type(16))) __bf16 v16bf;
typedef __attribute__((ext_vector_type(8)))  __bf16 v8bf;
typedef __attribute__((ext_vector_type(8)))  float  v8f;

constexpr int H_DIM   = 2048;     // hidden
constexpr int I_DIM   = 16384;    // intermediate
constexpr int T_TOK   = 8192;     // B*S
constexpr int M_TILE  = 16;       // tokens per workgroup
constexpr int CHUNK   = 128;      // intermediate columns per iteration
constexpr int NTHREADS = 512;     // 16 waves of 32
constexpr int XSTRIDE = H_DIM + 8;   // bf16 elems; 4112B row stride (16B aligned, bank-skewed)
constexpr int HSTRIDE = CHUNK + 8;   // 272B row stride
constexpr int GSTRIDE = CHUNK + 4;   // fp32 staging stride
constexpr int NEXP    = 5;
constexpr int LORA_R  = 16;
constexpr float LORA_SCALE = 2.0f;

__device__ __forceinline__ float gelu_tanh(float x) {
    float x3 = x * x * x;
    return 0.5f * x * (1.0f + tanhf(0.7978845608028654f * (x + 0.044715f * x3)));
}

// A fragment (16x32 bf16, MxK) from LDS row-major tile.
// Lane L holds row M=L&15; lanes 0-15 cover K[0..15],
// lanes 16-31 cover K[16..31] -> two contiguous 16B LDS reads.
__device__ __forceinline__ v16bf load_a_frag_lds(const __bf16* base, int stride, int lane) {
    int m  = lane & 15;
    int kh = lane >> 4;
    const __bf16* p = base + m * stride + kh * 16;
    union { v16bf v; v8bf h[2]; } u;
    u.h[0] = *(const v8bf*)(p);        // K = kh*16 .. kh*16+7
    u.h[1] = *(const v8bf*)(p + 8);    // K = kh*16+8 .. kh*16+15
    return u.v;
}

// B fragment (32x16 bf16, KxN) fetched directly from global fp32 and converted.
// Lane L holds column N=L&15; lanes 0-15 carry K 0..15, lanes 16-31 K 16..31
// -> 16 consecutive fp32 (64B) per lane from the weight row (row-major [N][K]).
__device__ __forceinline__ v16bf load_b_frag_global(const float* rowbase, int lane) {
    int kh = lane >> 4;
    const float4* p = (const float4*)(rowbase + kh * 16);
    float4 f0 = p[0], f1 = p[1], f2 = p[2], f3 = p[3];
    union { v16bf v; __bf16 e[16]; } u;
    u.e[0]  = (__bf16)f0.x; u.e[1]  = (__bf16)f0.y; u.e[2]  = (__bf16)f0.z; u.e[3]  = (__bf16)f0.w;
    u.e[4]  = (__bf16)f1.x; u.e[5]  = (__bf16)f1.y; u.e[6]  = (__bf16)f1.z; u.e[7]  = (__bf16)f1.w;
    u.e[8]  = (__bf16)f2.x; u.e[9]  = (__bf16)f2.y; u.e[10] = (__bf16)f2.z; u.e[11] = (__bf16)f2.w;
    u.e[12] = (__bf16)f3.x; u.e[13] = (__bf16)f3.y; u.e[14] = (__bf16)f3.z; u.e[15] = (__bf16)f3.w;
    return u.v;
}

// ---------------------------------------------------------------------------
// Kernel 1: router. One wave per token: logits = x . router_w[e], argmax,
// softmax, atomic accumulation of prob/count sums for the aux loss.
// ---------------------------------------------------------------------------
__global__ __launch_bounds__(256) void router_kernel(
    const float* __restrict__ x, const float* __restrict__ rw,
    int* __restrict__ sel_out, float* __restrict__ prob_acc,
    float* __restrict__ cnt_acc)
{
    int wave  = threadIdx.x >> 5;
    int lane  = threadIdx.x & 31;
    int token = blockIdx.x * 8 + wave;
    if (token >= T_TOK) return;

    float acc[NEXP];
#pragma unroll
    for (int e = 0; e < NEXP; ++e) acc[e] = 0.0f;

    const float* xr = x + (size_t)token * H_DIM;
    for (int h = lane; h < H_DIM; h += 32) {
        float xv = xr[h];
#pragma unroll
        for (int e = 0; e < NEXP; ++e) acc[e] += xv * rw[e * H_DIM + h];
    }
#pragma unroll
    for (int e = 0; e < NEXP; ++e)
#pragma unroll
        for (int off = 16; off > 0; off >>= 1)
            acc[e] += __shfl_xor(acc[e], off, 32);

    if (lane == 0) {
        float mx = acc[0]; int am = 0;
#pragma unroll
        for (int e = 1; e < NEXP; ++e) if (acc[e] > mx) { mx = acc[e]; am = e; }
        float p[NEXP], s = 0.0f;
#pragma unroll
        for (int e = 0; e < NEXP; ++e) { p[e] = __expf(acc[e] - mx); s += p[e]; }
        float inv = 1.0f / s;
#pragma unroll
        for (int e = 0; e < NEXP; ++e) atomicAdd(&prob_acc[e], p[e] * inv);
        atomicAdd(&cnt_acc[am], 1.0f);
        sel_out[token] = am;
    }
}

// ---------------------------------------------------------------------------
// Kernel 2: fused GeGLU MLP + LoRA epilogue. 16 tokens per workgroup.
// Waves 0-7: gate tiles; waves 8-15: up tiles; all 16 waves: down accumulation
// into persistent VGPR accumulators (each wave owns 128 output columns).
// ---------------------------------------------------------------------------
__global__ __launch_bounds__(NTHREADS) void moe_mlp_kernel(
    const float* __restrict__ x,      const float* __restrict__ gate_w,
    const float* __restrict__ up_w,   const float* __restrict__ down_w,
    const float* __restrict__ lora_A, const float* __restrict__ lora_B,
    const int*   __restrict__ sel,    float* __restrict__ out)
{
    __shared__ __attribute__((aligned(16))) __bf16 lds_x[M_TILE * XSTRIDE];
    __shared__ __attribute__((aligned(16))) __bf16 lds_h[M_TILE * HSTRIDE];
    __shared__ __attribute__((aligned(16))) float  lds_g[M_TILE * GSTRIDE];
    __shared__ __attribute__((aligned(16))) float  lds_u[M_TILE * GSTRIDE];
    __shared__ __attribute__((aligned(16))) float  lds_t[M_TILE * LORA_R];

    const int tid  = threadIdx.x;
    const int wave = tid >> 5;
    const int lane = tid & 31;
    const int tok0 = blockIdx.x * M_TILE;

    // Stage x tile (fp32 -> bf16) into LDS once; reused for all 128 I-chunks.
    for (int idx = tid; idx < M_TILE * H_DIM; idx += NTHREADS) {
        int m = idx / H_DIM, h = idx % H_DIM;
        lds_x[m * XSTRIDE + h] = (__bf16)x[(size_t)(tok0 + m) * H_DIM + h];
    }
    __syncthreads();

    // Phase 0: low-rank LoRA intermediate t[m][r] for each token's expert.
    if (tid < M_TILE * LORA_R) {
        int m = tid >> 4, r = tid & 15;
        int e = sel[tok0 + m] - 1;
        float t = 0.0f;
        if (e >= 0) {
            const float* la = lora_A + ((size_t)e * LORA_R + r) * H_DIM;
            for (int h = 0; h < H_DIM; ++h)
                t += (float)lds_x[m * XSTRIDE + h] * la[h];
        }
        lds_t[tid] = t;
    }

    // Persistent output accumulators: wave owns columns [wave*128, wave*128+128).
    v8f acc[8];
#pragma unroll
    for (int i = 0; i < 8; ++i) acc[i] = v8f{};

    const float* W    = (wave < 8) ? gate_w : up_w;
    const int    tcol = (wave & 7) * 16;

    for (int chunk = 0; chunk < I_DIM / CHUNK; ++chunk) {
        const int ibase = chunk * CHUNK;

        // ---- Phase 1: 16x16 gate/up tile, K-reduction over H ----
        v8f c = v8f{};
        const float* rowW = W + (size_t)(ibase + tcol + (lane & 15)) * H_DIM;
        if (chunk + 1 < I_DIM / CHUNK)
            __builtin_prefetch(rowW + (size_t)CHUNK * H_DIM, 0, 0);
#pragma unroll 4
        for (int ks = 0; ks < H_DIM; ks += 32) {
            v16bf a = load_a_frag_lds(lds_x + ks, XSTRIDE, lane);
            v16bf b = load_b_frag_global(rowW + ks, lane);
            c = __builtin_amdgcn_wmma_f32_16x16x32_bf16(
                    false, a, false, b, (short)0, c, false, false);
        }
        {   // C layout: VGPR v -> row v + 8*(lane>>4), col lane&15
            float* dst = (wave < 8) ? lds_g : lds_u;
            int kh = lane >> 4, nc = lane & 15;
#pragma unroll
            for (int v = 0; v < 8; ++v)
                dst[(v + 8 * kh) * GSTRIDE + tcol + nc] = c[v];
        }
        __syncthreads();

        // ---- GeGLU elementwise: h = gelu(g) * u, bf16 into lds_h ----
        for (int idx = tid; idx < M_TILE * CHUNK; idx += NTHREADS) {
            int m = idx >> 7, col = idx & (CHUNK - 1);
            float g = lds_g[m * GSTRIDE + col];
            float u = lds_u[m * GSTRIDE + col];
            lds_h[m * HSTRIDE + col] = (__bf16)(gelu_tanh(g) * u);
        }
        __syncthreads();

        // ---- Phase 2: out[16, wave's 128 cols] += h[16,128] @ down^T ----
#pragma unroll
        for (int nt = 0; nt < 8; ++nt) {
            int n = wave * 128 + nt * 16 + (lane & 15);
            const float* drow = down_w + (size_t)n * I_DIM + ibase;
#pragma unroll
            for (int ks = 0; ks < CHUNK; ks += 32) {
                v16bf a = load_a_frag_lds(lds_h + ks, HSTRIDE, lane);
                v16bf b = load_b_frag_global(drow + ks, lane);
                acc[nt] = __builtin_amdgcn_wmma_f32_16x16x32_bf16(
                              false, a, false, b, (short)0, acc[nt], false, false);
            }
        }
        __syncthreads();   // protect lds_g/lds_u/lds_h for next chunk
    }

    // ---- Epilogue: add LoRA correction, write fp32 output ----
    {
        int kh = lane >> 4, nc = lane & 15;
#pragma unroll
        for (int nt = 0; nt < 8; ++nt) {
            int n = wave * 128 + nt * 16 + nc;
#pragma unroll
            for (int v = 0; v < 8; ++v) {
                int m = v + 8 * kh;
                float val = acc[nt][v];
                int e = sel[tok0 + m] - 1;
                if (e >= 0) {
                    const float* lb = lora_B + ((size_t)e * H_DIM + n) * LORA_R;
                    float lr = 0.0f;
#pragma unroll
                    for (int r = 0; r < LORA_R; ++r)
                        lr += lds_t[m * LORA_R + r] * lb[r];
                    val += LORA_SCALE * lr;
                }
                out[(size_t)(tok0 + m) * H_DIM + n] = val;
            }
        }
    }
}

// ---------------------------------------------------------------------------
// Kernel 3: aux load-balancing loss.
// ---------------------------------------------------------------------------
__global__ void aux_kernel(const float* __restrict__ prob_acc,
                           const float* __restrict__ cnt_acc,
                           float* __restrict__ aux_out)
{
    if (threadIdx.x == 0 && blockIdx.x == 0) {
        const float invT = 1.0f / (float)T_TOK;
        float s = 0.0f;
#pragma unroll
        for (int e = 0; e < NEXP; ++e)
            s += (prob_acc[e] * invT) * (cnt_acc[e] * invT);
        *aux_out = s * (float)NEXP * 0.01f;
    }
}

extern "C" void kernel_launch(void* const* d_in, const int* in_sizes, int n_in,
                              void* d_out, int out_size, void* d_ws, size_t ws_size,
                              hipStream_t stream)
{
    const float* x  = (const float*)d_in[0];   // [4,2048,2048]
    const float* rw = (const float*)d_in[1];   // [5,2048]
    const float* gw = (const float*)d_in[2];   // [16384,2048]
    const float* uw = (const float*)d_in[3];   // [16384,2048]
    const float* dw = (const float*)d_in[4];   // [2048,16384]
    const float* lA = (const float*)d_in[5];   // [4,16,2048]
    const float* lB = (const float*)d_in[6];   // [4,2048,16]
    float* out = (float*)d_out;

    float* prob_acc = (float*)d_ws;            // [5]
    float* cnt_acc  = prob_acc + NEXP;         // [5]
    int*   sel      = (int*)((char*)d_ws + 64);// [8192]

    hipMemsetAsync(d_ws, 0, 64, stream);
    router_kernel<<<T_TOK / 8, 256, 0, stream>>>(x, rw, sel, prob_acc, cnt_acc);
    moe_mlp_kernel<<<T_TOK / M_TILE, NTHREADS, 0, stream>>>(
        x, gw, uw, dw, lA, lB, sel, out);
    aux_kernel<<<1, 1, 0, stream>>>(prob_acc, cnt_acc,
                                    out + (size_t)T_TOK * H_DIM);
}
